// FeedBack_17368847745190
// MI455X (gfx1250) — compile-verified, matrix operand order checked by reading, not run on hardware
//
#include <hip/hip_runtime.h>

// LSTM feedback model for MI455X (gfx1250), wave32 + WMMA.
// 128 blocks x 256 threads; block = 16 batch rows; wave w = units [16w,16w+16).
// rec_kernel held as f16 WMMA B-fragments in VGPRs for the whole kernel.
// h state exchanged via 4KB LDS per step; c state in VGPRs.
// Activations via branch-free hardware exp2/rcp (latency-bound recurrence:
// the per-step elementwise tail sits on the critical path of 319 serial steps).

typedef __attribute__((ext_vector_type(16))) _Float16 v16h_t;
typedef __attribute__((ext_vector_type(8)))  _Float16 v8h_t;
typedef __attribute__((ext_vector_type(8)))  float    v8f_t;

namespace {
constexpr int kUnits    = 128;
constexpr int kGateCols = 512;   // 4 * UNITS
constexpr int kB        = 2048;
constexpr int kT        = 256;
constexpr int kOutSteps = 64;
constexpr int kRows     = 16;    // batch rows per block
constexpr int kThreads  = 256;   // 8 waves of 32

union V16H { v16h_t v; _Float16 e[16]; };
union V8F  { v8f_t  v; float    e[8];  };

// Branch-free activations on the hardware transcendental pipe.
// v_exp_f32 is exp2; saturation (inf/0) gives the right limits for gates.
constexpr float kLog2e  = 1.4426950408889634f;

__device__ inline float fast_sigmoid(float x) {
  float e = __builtin_amdgcn_exp2f(-x * kLog2e);        // exp(-x)
  return __builtin_amdgcn_rcpf(1.0f + e);               // 1/(1+exp(-x))
}
__device__ inline float fast_tanh(float x) {
  float e = __builtin_amdgcn_exp2f(-2.0f * kLog2e * x); // exp(-2x)
  return 2.0f * __builtin_amdgcn_rcpf(1.0f + e) - 1.0f; // 2*sigmoid(2x)-1
}
} // namespace

__global__ __launch_bounds__(kThreads) void lstm_feedback_kernel(
    const float* __restrict__ inputs,      // [B, T, 1]
    const float* __restrict__ kernel,      // [1, 512]
    const float* __restrict__ rec_kernel,  // [128, 512] row-major
    const float* __restrict__ bias,        // [512]
    const float* __restrict__ dense_w,     // [128, 1]
    const float* __restrict__ dense_b,     // [1]
    float* __restrict__ out)               // [B, 64, 1]
{
  __shared__ _Float16 h_lds[kRows * kUnits];  // h state, row-major [row][unit], f16
  __shared__ float    x_lds[kRows];           // per-row scalar input / fed-back pred
  __shared__ float    dw_lds[kUnits];         // dense_w

  const int tid     = threadIdx.x;
  const int wave    = tid >> 5;
  const int lane    = tid & 31;
  const int lane16  = lane & 15;
  const int hi_half = (lane >> 4) & 1;          // lanes 16-31 -> 1
  const int unit    = wave * 16 + lane16;       // unit column owned by this lane
  const int row0    = blockIdx.x * kRows;

  // ---------------- one-time setup ----------------
  for (int i = tid; i < kRows * kUnits; i += kThreads) h_lds[i] = (_Float16)0.0f;
  if (tid < kUnits) dw_lds[tid] = dense_w[tid];

  // per-lane input-kernel and bias constants, one per gate
  float kc[4], bc[4];
  #pragma unroll
  for (int gi = 0; gi < 4; ++gi) {
    kc[gi] = kernel[gi * kUnits + unit];
    bc[gi] = bias[gi * kUnits + unit];
  }
  const float db = dense_b[0];

  // Recurrent weights as WMMA B fragments (K=32 x N=16), register-resident.
  // bfrag[gi][kt]: K = 32*kt .. 32*kt+31, N column = gi*128 + unit.
  // Lane L holds column N = L%16; per-lane K striping mirrors the ISA 16-bit
  // operand layout: VGPR v, half p -> klocal = (v&4?16:0) + (L>=16?8:0) + 2*(v&3) + p.
  V16H bfrag[4][4];
  #pragma unroll
  for (int gi = 0; gi < 4; ++gi) {
    const int col = gi * kUnits + unit;
    #pragma unroll
    for (int kt = 0; kt < 4; ++kt) {
      #pragma unroll
      for (int hh = 0; hh < 16; ++hh) {
        const int v = hh >> 1, p = hh & 1;
        const int klocal = ((v & 4) ? 16 : 0) + (hi_half ? 8 : 0) + 2 * (v & 3) + p;
        bfrag[gi][kt].e[hh] =
            (_Float16)rec_kernel[(kt * 32 + klocal) * kGateCols + col];
      }
    }
  }

  float creg[8];
  #pragma unroll
  for (int r = 0; r < 8; ++r) creg[r] = 0.0f;

  __syncthreads();  // h_lds zeroed, dw_lds ready

  // A fragment (16x32 f16) for k-tile kt from row-major h_lds.
  // Lane L: row = L%16; halves 0-7 = units base..base+7, halves 8-15 = base+16..base+23,
  // base = 32*kt + (L>=16 ? 8 : 0)  -> two contiguous 16B LDS loads.
  auto load_a = [&](int kt) -> v16h_t {
    const int base = lane16 * kUnits + kt * 32 + hi_half * 8;
    const v8h_t lo = *(const v8h_t*)&h_lds[base];
    const v8h_t hi = *(const v8h_t*)&h_lds[base + 16];
    union { v16h_t v; v8h_t h[2]; } u;
    u.h[0] = lo; u.h[1] = hi;
    return u.v;
  };

  // One LSTM cell step: 16 WMMAs (4 gates x 4 k-tiles) + gate math + h store.
  auto cell = [&](const v16h_t a[4]) {
    V8F z[4];
    #pragma unroll
    for (int gi = 0; gi < 4; ++gi) {
      v8f_t acc = {};
      #pragma unroll
      for (int kt = 0; kt < 4; ++kt) {
        acc = __builtin_amdgcn_wmma_f32_16x16x32_f16(
            /*neg_a=*/false, a[kt], /*neg_b=*/false, bfrag[gi][kt].v,
            /*c_mod=*/(short)0, acc, /*reuse_a=*/false, /*reuse_b=*/false);
      }
      z[gi].v = acc;
    }
    // C/D layout: reg r, lane L -> (row = (L>=16?8:0)+r, unit = L%16 (+16w global))
    #pragma unroll
    for (int r = 0; r < 8; ++r) {
      const int row = hi_half * 8 + r;
      const float x  = x_lds[row];
      const float zi = z[0].e[r] + x * kc[0] + bc[0];
      const float zf = z[1].e[r] + x * kc[1] + bc[1];
      const float zg = z[2].e[r] + x * kc[2] + bc[2];
      const float zo = z[3].e[r] + x * kc[3] + bc[3];
      const float ig = fast_sigmoid(zi);
      const float fg = fast_sigmoid(zf);
      const float gg = fast_tanh(zg);
      const float og = fast_sigmoid(zo);
      const float c  = fg * creg[r] + ig * gg;
      creg[r] = c;
      const float h  = og * fast_tanh(c);
      h_lds[row * kUnits + unit] = (_Float16)h;
    }
  };

  // Dense head for 16 rows (wave 0, lanes 0-15); writes out + feeds back x.
  // Vectorized: 16 x ds_load_b128 instead of 128 scalar u16 loads.
  auto predict = [&](int step) {
    if (tid < kRows) {
      float p = db;
      #pragma unroll
      for (int u0 = 0; u0 < kUnits; u0 += 8) {
        const v8h_t hv = *(const v8h_t*)&h_lds[tid * kUnits + u0];
        #pragma unroll
        for (int j = 0; j < 8; ++j) p += (float)hv[j] * dw_lds[u0 + j];
      }
      out[(row0 + tid) * kOutSteps + step] = p;
      x_lds[tid] = p;
    }
  };

  // ---------------- warmup: 256 steps over the input window ----------------
  for (int t = 0; t < kT; ++t) {
    v16h_t a[4];
    #pragma unroll
    for (int kt = 0; kt < 4; ++kt) a[kt] = load_a(kt);
    if (tid < kRows) x_lds[tid] = inputs[(row0 + tid) * kT + t];
    __syncthreads();  // x ready; A reads done -> safe to overwrite h below
    cell(a);
    __syncthreads();  // new h visible
  }

  predict(0);

  // ---------------- autoregressive decode: 63 more steps ----------------
  for (int s = 1; s < kOutSteps; ++s) {
    v16h_t a[4];
    #pragma unroll
    for (int kt = 0; kt < 4; ++kt) a[kt] = load_a(kt);
    __syncthreads();  // predict's x_lds write visible; A reads done
    cell(a);
    __syncthreads();  // new h visible for predict
    predict(s);
  }
}

extern "C" void kernel_launch(void* const* d_in, const int* in_sizes, int n_in,
                              void* d_out, int out_size, void* d_ws, size_t ws_size,
                              hipStream_t stream) {
  (void)in_sizes; (void)n_in; (void)d_ws; (void)ws_size; (void)out_size;
  const float* inputs     = (const float*)d_in[0];
  const float* kernel_    = (const float*)d_in[1];
  const float* rec_kernel = (const float*)d_in[2];
  const float* bias       = (const float*)d_in[3];
  const float* dense_w    = (const float*)d_in[4];
  const float* dense_b    = (const float*)d_in[5];
  float* out = (float*)d_out;

  hipLaunchKernelGGL(lstm_feedback_kernel,
                     dim3(kB / kRows), dim3(kThreads), 0, stream,
                     inputs, kernel_, rec_kernel, bias, dense_w, dense_b, out);
}
